// Decoder_12790412608085
// MI455X (gfx1250) — matrix operand order repaired and checked by reading, not compiled, not run on hardware
//
#include <hip/hip_runtime.h>
#include <hip/hip_bf16.h>

// ---------------- problem constants ----------------
#define Hdim   128
#define Tlen   82
#define Bsz    32
#define G3     384            // 3*H
#define VOC    32000
#define ROWS   (Bsz * Tlen)   // 2624 = 164 * 16
#define RT     164            // row tiles of 16
#define NT_V   (VOC / 16)     // 2000 vocab tiles
#define NP_V   (NT_V / 2)     // 1000 vocab tile-pairs

typedef __attribute__((ext_vector_type(16))) __bf16 v16bf;
typedef __attribute__((ext_vector_type(8)))  float  v8f;

union ABf { v16bf v; unsigned u[8]; };

// ---------------- helpers ----------------
__device__ __forceinline__ unsigned short f2bf(float f) {
    unsigned u = __float_as_uint(f);
    unsigned r = (u + 0x7FFFu + ((u >> 16) & 1u)) >> 16;   // RNE
    return (unsigned short)r;
}
__device__ __forceinline__ unsigned pack2(float a, float b) {
    return (unsigned)f2bf(a) | ((unsigned)f2bf(b) << 16);
}

// A-matrix 16x32 bf16 fragment from packed-bf16 LDS tile [16 rows][64 uints]
// ISA 7.12.2: lane m (0-15) / m+16: vgpr v -> K = (v>>2)*16 + half*8 + 2*(v&3)
__device__ __forceinline__ v16bf load_a_frag(const unsigned* xs, int kc) {
    int lane = threadIdx.x & 31;
    int m    = lane & 15;
    int half = lane >> 4;
    const unsigned* row = xs + m * 64 + (kc >> 1);
    ABf a;
#pragma unroll
    for (int v = 0; v < 8; ++v) {
        int k = (v >> 2) * 16 + half * 8 + 2 * (v & 3);
        a.u[v] = row[k >> 1];
    }
    return a.v;
}

// B-matrix 32x16 bf16 fragment. B[k][n] = W[n][k]; W packed bf16, 64 uints/row.
// lane n (0-15) / n+16: vgpr v -> K = (v>>2)*16 + half*8 + 2*(v&3)
__device__ __forceinline__ v16bf load_b_frag(const unsigned* __restrict__ Wbf,
                                             int ncol_base, int kc) {
    int lane  = threadIdx.x & 31;
    int n     = lane & 15;
    int half  = lane >> 4;
    const unsigned* row = Wbf + (long)(ncol_base + n) * 64 + (kc >> 1);
    ABf b;
#pragma unroll
    for (int v = 0; v < 8; ++v) b.u[v] = row[(v >> 2) * 8 + half * 4 + (v & 3)];
    return b.v;
}

__device__ __forceinline__ v8f wmma_bf16(v16bf a, v16bf b, v8f c) {
    return __builtin_amdgcn_wmma_f32_16x16x32_bf16(false, a, false, b,
                                                   (short)0, c, false, false);
}

// ---------------- k0: f32 -> packed bf16 weight conversion ----------------
__global__ void k0_convert(const float* __restrict__ src,
                           unsigned* __restrict__ dst, int n_pairs) {
    int i = blockIdx.x * blockDim.x + threadIdx.x;
    if (i < n_pairs) dst[i] = pack2(src[2 * i], src[2 * i + 1]);
}

// ---------------- k1: x = relu(emb[tok]); x_proj = x @ W_ih^T + b_ih -------
__global__ __launch_bounds__(256) void k1_xproj(
    const int* __restrict__ target, const float* __restrict__ emb,
    const unsigned* __restrict__ Wih_bf, const float* __restrict__ b_ih,
    float* __restrict__ xproj) {
    __shared__ unsigned xs[16 * 64];      // 16 rows x 128 bf16 packed
    int rb  = blockIdx.x;
    int tid = threadIdx.x;
    for (int idx = tid; idx < 1024; idx += 256) {
        int i = idx >> 6, ku = idx & 63;
        int row = rb * 16 + i;
        int bb = row / Tlen, t = row % Tlen;
        int tok = (t == 0) ? 0 : target[bb * Tlen + t - 1];
        const float* e = emb + (long)tok * Hdim + ku * 2;
        xs[i * 64 + ku] = pack2(fmaxf(e[0], 0.f), fmaxf(e[1], 0.f));
    }
    __syncthreads();
    int wave = tid >> 5, lane = tid & 31;
    int n = lane & 15, mh = lane >> 4;
    v16bf afr[4];
#pragma unroll
    for (int kc = 0; kc < 4; ++kc) afr[kc] = load_a_frag(xs, kc * 32);
#pragma unroll
    for (int j = 0; j < 3; ++j) {
        int ntile = wave * 3 + j;                      // 24 tiles / 8 waves
        v8f acc = {};
#pragma unroll
        for (int kc = 0; kc < 4; ++kc)
            acc = wmma_bf16(afr[kc], load_b_frag(Wih_bf, ntile * 16, kc * 32), acc);
        float bias = b_ih[ntile * 16 + n];
#pragma unroll
        for (int r = 0; r < 8; ++r) {
            int m = mh * 8 + r;
            xproj[(long)(rb * 16 + m) * G3 + ntile * 16 + n] = acc[r] + bias;
        }
    }
}

// ---------------- k2: sequential GRU scan (one persistent block) ----------
__global__ __launch_bounds__(512) void k2_scan(
    const float* __restrict__ h0, const unsigned* __restrict__ Whh_bf,
    const float* __restrict__ b_hh, const float* __restrict__ xproj,
    unsigned* __restrict__ hs_bf, float* __restrict__ hT_out) {
    __shared__ float    hsh[32 * 128];    // h (fp32 master)     16 KB
    __shared__ unsigned hbf[32 * 64];     // h packed bf16        8 KB
    __shared__ float    hp[32 * 384];     // h @ W_hh^T + b_hh   48 KB
    __shared__ float    bhh[384];
    int tid = threadIdx.x;
    for (int i = tid; i < 32 * 128; i += 512) hsh[i] = h0[i];
    for (int i = tid; i < 384; i += 512) bhh[i] = b_hh[i];
    __syncthreads();
    int wave = tid >> 5, lane = tid & 31;
    int n = lane & 15, mh = lane >> 4;
    for (int t = 0; t < Tlen; ++t) {
        for (int idx = tid; idx < 32 * 64; idx += 512)
            hbf[idx] = pack2(hsh[idx * 2], hsh[idx * 2 + 1]);
        __syncthreads();
#pragma unroll
        for (int j = 0; j < 3; ++j) {                  // 48 tiles / 16 waves
            int tile = wave * 3 + j;
            int ntile = tile % 24, mtile = tile / 24;
            v8f acc = {};
#pragma unroll
            for (int kc = 0; kc < 4; ++kc) {
                v16bf a = load_a_frag(hbf + mtile * 16 * 64, kc * 32);
                acc = wmma_bf16(a, load_b_frag(Whh_bf, ntile * 16, kc * 32), acc);
            }
            float bias = bhh[ntile * 16 + n];
#pragma unroll
            for (int r = 0; r < 8; ++r) {
                int m = mtile * 16 + mh * 8 + r;
                hp[m * 384 + ntile * 16 + n] = acc[r] + bias;
            }
        }
        __syncthreads();
        {   // gates: 4096 elements, 8 contiguous per thread
            int base = tid * 8;
            int m = base >> 7, k0 = base & 127;
            const float* xp = xproj + ((long)m * Tlen + t) * G3;
            unsigned packs[4];
#pragma unroll
            for (int q = 0; q < 8; q += 2) {
                float hn2[2];
#pragma unroll
                for (int s = 0; s < 2; ++s) {
                    int k = k0 + q + s;
                    float hr = hp[m * 384 + k];
                    float hz = hp[m * 384 + 128 + k];
                    float hn = hp[m * 384 + 256 + k];
                    float rg = 1.f / (1.f + __expf(-(xp[k] + hr)));
                    float zg = 1.f / (1.f + __expf(-(xp[128 + k] + hz)));
                    float ng = tanhf(xp[256 + k] + rg * hn);
                    float hnew = (1.f - zg) * ng + zg * hsh[m * 128 + k];
                    hsh[m * 128 + k] = hnew;
                    hn2[s] = hnew;
                }
                packs[q >> 1] = pack2(hn2[0], hn2[1]);
            }
            unsigned* dst = hs_bf + ((long)m * Tlen + t) * 64 + (k0 >> 1);
#pragma unroll
            for (int q = 0; q < 4; ++q) dst[q] = packs[q];
        }
        __syncthreads();
    }
    for (int i = tid; i < 32 * 128; i += 512) hT_out[i] = hsh[i];
}

// ---------------- k3: logits = hs @ W_out^T + b_out ; log_softmax ----------
// Two vocab tiles per iteration => two independent WMMA accumulator chains
// per wave (hides WMMA->WMMA RAW latency, §7.12.1) and doubled load clauses.
__global__ __launch_bounds__(512) void k3_logits(
    const unsigned* __restrict__ hs_bf, const unsigned* __restrict__ Wout_bf,
    const float* __restrict__ b_out, float* __restrict__ out) {
    __shared__ unsigned xs[16 * 64];
    __shared__ float wmax[16 * 16], wsum[16 * 16], lse[16];
    int rb = blockIdx.x, tid = threadIdx.x;
    for (int idx = tid; idx < 1024; idx += 512)
        xs[idx] = hs_bf[(long)rb * 1024 + idx];
    __syncthreads();
    int wave = tid >> 5, lane = tid & 31;
    int n = lane & 15, mh = lane >> 4;
    v16bf afr[4];
#pragma unroll
    for (int kc = 0; kc < 4; ++kc) afr[kc] = load_a_frag(xs, kc * 32);

    // ---- phase A: per-row online max / sum-exp over all 2000 tiles ----
    float rmax[8], rsum[8];
#pragma unroll
    for (int r = 0; r < 8; ++r) { rmax[r] = -3.0e38f; rsum[r] = 0.f; }
    for (int p = wave; p < NP_V; p += 16) {
        int nt0 = 2 * p, nt1 = 2 * p + 1;
        v8f acc0 = {}, acc1 = {};
#pragma unroll
        for (int kc = 0; kc < 4; ++kc) {
            v16bf b0 = load_b_frag(Wout_bf, nt0 * 16, kc * 32);
            v16bf b1 = load_b_frag(Wout_bf, nt1 * 16, kc * 32);
            acc0 = wmma_bf16(afr[kc], b0, acc0);
            acc1 = wmma_bf16(afr[kc], b1, acc1);
        }
        float bias0 = b_out[nt0 * 16 + n];
        float bias1 = b_out[nt1 * 16 + n];
#pragma unroll
        for (int r = 0; r < 8; ++r) {
            float v0 = acc0[r] + bias0;
            float v1 = acc1[r] + bias1;
            float om = rmax[r];
            if (v0 > om) { rsum[r] = rsum[r] * __expf(om - v0) + 1.f; rmax[r] = v0; }
            else         { rsum[r] += __expf(v0 - om); }
            om = rmax[r];
            if (v1 > om) { rsum[r] = rsum[r] * __expf(om - v1) + 1.f; rmax[r] = v1; }
            else         { rsum[r] += __expf(v1 - om); }
        }
    }
    // reduce across the 16 lanes of each half (rows 0-7 / 8-15)
#pragma unroll
    for (int off = 1; off < 16; off <<= 1) {
#pragma unroll
        for (int r = 0; r < 8; ++r) {
            float om = __shfl_xor(rmax[r], off, 32);
            float os = __shfl_xor(rsum[r], off, 32);
            float M = fmaxf(rmax[r], om);
            rsum[r] = rsum[r] * __expf(rmax[r] - M) + os * __expf(om - M);
            rmax[r] = M;
        }
    }
    if (n == 0) {
#pragma unroll
        for (int r = 0; r < 8; ++r) {
            int row = mh * 8 + r;
            wmax[row * 16 + wave] = rmax[r];
            wsum[row * 16 + wave] = rsum[r];
        }
    }
    __syncthreads();
    if (tid < 16) {
        float M = -3.0e38f;
        for (int w = 0; w < 16; ++w) M = fmaxf(M, wmax[tid * 16 + w]);
        float S = 0.f;
        for (int w = 0; w < 16; ++w) S += wsum[tid * 16 + w] * __expf(wmax[tid * 16 + w] - M);
        lse[tid] = M + __logf(S);
    }
    __syncthreads();
    float mylse[8];
#pragma unroll
    for (int r = 0; r < 8; ++r) mylse[r] = lse[mh * 8 + r];

    // ---- phase B: recompute (W_out hot in L2) and stream results out ----
    for (int p = wave; p < NP_V; p += 16) {
        int nt0 = 2 * p, nt1 = 2 * p + 1;
        v8f acc0 = {}, acc1 = {};
#pragma unroll
        for (int kc = 0; kc < 4; ++kc) {
            v16bf b0 = load_b_frag(Wout_bf, nt0 * 16, kc * 32);
            v16bf b1 = load_b_frag(Wout_bf, nt1 * 16, kc * 32);
            acc0 = wmma_bf16(afr[kc], b0, acc0);
            acc1 = wmma_bf16(afr[kc], b1, acc1);
        }
        float bias0 = b_out[nt0 * 16 + n];
        float bias1 = b_out[nt1 * 16 + n];
#pragma unroll
        for (int r = 0; r < 8; ++r) {
            int m = mh * 8 + r;
            float* rowp = out + (long)(rb * 16 + m) * VOC;
            __builtin_nontemporal_store(acc0[r] + bias0 - mylse[r],
                                        rowp + nt0 * 16 + n);
            __builtin_nontemporal_store(acc1[r] + bias1 - mylse[r],
                                        rowp + nt1 * 16 + n);
        }
    }
}

// ---------------- launcher ----------------
extern "C" void kernel_launch(void* const* d_in, const int* in_sizes, int n_in,
                              void* d_out, int out_size, void* d_ws, size_t ws_size,
                              hipStream_t stream) {
    const float* enc_hidden = (const float*)d_in[1];   // [1,B,H]
    const int*   target     = (const int*)d_in[2];     // [B,T]
    const float* emb        = (const float*)d_in[3];   // [V,H]
    const float* W_ih       = (const float*)d_in[4];   // [3H,H]
    const float* W_hh       = (const float*)d_in[5];
    const float* b_ih       = (const float*)d_in[6];
    const float* b_hh       = (const float*)d_in[7];
    const float* W_out      = (const float*)d_in[8];   // [V,H]
    const float* b_out      = (const float*)d_in[9];
    float* out = (float*)d_out;

    // workspace layout (256B aligned)
    char* ws = (char*)d_ws;
    unsigned* Wout_bf = (unsigned*)(ws);                       // 8,192,000 B
    unsigned* Wih_bf  = (unsigned*)(ws + 8192000);             //    98,304 B
    unsigned* Whh_bf  = (unsigned*)(ws + 8290304);             //    98,304 B
    float*    xproj   = (float*)   (ws + 8388608);             // 4,030,464 B
    unsigned* hs_bf   = (unsigned*)(ws + 12419072);            //   671,744 B

    // k0: weight conversion to packed bf16
    k0_convert<<<(VOC * Hdim / 2 + 255) / 256, 256, 0, stream>>>(W_out, Wout_bf, VOC * Hdim / 2);
    k0_convert<<<(G3 * Hdim / 2 + 255) / 256, 256, 0, stream>>>(W_ih, Wih_bf, G3 * Hdim / 2);
    k0_convert<<<(G3 * Hdim / 2 + 255) / 256, 256, 0, stream>>>(W_hh, Whh_bf, G3 * Hdim / 2);

    // k1: embedding + relu + input-side gate projections
    k1_xproj<<<RT, 256, 0, stream>>>(target, emb, Wih_bf, b_ih, xproj);

    // k2: sequential GRU scan (single persistent workgroup)
    k2_scan<<<1, 512, 0, stream>>>(enc_hidden, Whh_bf, b_hh, xproj, hs_bf,
                                   out + (long)Bsz * Tlen * VOC);

    // k3: vocab projection + log_softmax (recompute-normalize, NT stores)
    k3_logits<<<RT, 512, 0, stream>>>(hs_bf, Wout_bf, b_out, out);
}
